// MultiheadDiffAttn_17952963297574
// MI455X (gfx1250) — matrix operand order repaired and checked by reading, not compiled, not run on hardware
//
#include <hip/hip_runtime.h>
#include <hip/hip_bf16.h>

// ---------------- problem constants ----------------
#define EMB     1024
#define NHEADS  16
#define T_DIM   2048
#define BATCH   2
#define HD      32          // differential half head dim (== WMMA K depth)
#define M_ROWS  (BATCH * T_DIM)   // 4096
#define LAMBDA_INIT_F 0.783605766532f
#define LOG2E_F 1.4426950408889634f
#define Q_SCALE 0.17677669529663687f   // 32^-0.5

typedef __bf16 v16bf __attribute__((ext_vector_type(16)));
typedef __bf16 v8bf  __attribute__((ext_vector_type(8)));
typedef float  v8f   __attribute__((ext_vector_type(8)));
typedef float  v4f   __attribute__((ext_vector_type(4)));
typedef unsigned int v2u __attribute__((ext_vector_type(2)));

__device__ __forceinline__ unsigned short f32_to_bf16(float f) {
    unsigned int u = __builtin_bit_cast(unsigned int, f);
    return (unsigned short)((u + 0x8000u) >> 16);   // round-half-up
}

// pack two f32 -> two bf16 in one dword (round-half-up)
__device__ __forceinline__ unsigned int pack2_bf16(float a, float b) {
    unsigned int ua = __builtin_bit_cast(unsigned int, a);
    unsigned int ub = __builtin_bit_cast(unsigned int, b);
    return ((ua + 0x8000u) >> 16) | ((ub + 0x8000u) & 0xffff0000u);
}

__device__ __forceinline__ v8f wmma_bf16(v16bf a, v16bf b, v8f c) {
    return __builtin_amdgcn_wmma_f32_16x16x32_bf16(
        false, a, false, b, (short)0, c, false, false);
}

__device__ __forceinline__ v16bf cat8(v8bf lo, v8bf hi) {
    return __builtin_shufflevector(lo, hi, 0,1,2,3,4,5,6,7,8,9,10,11,12,13,14,15);
}

// ---------------- fp32 -> bf16 cast (4 elems / thread) ----------------
__global__ void __launch_bounds__(256)
cast_f32_bf16(const float* __restrict__ in,
              unsigned short* __restrict__ out, int n) {
    int i = (blockIdx.x * blockDim.x + threadIdx.x) * 4;
    if (i < n) {
        v4f x = *(const v4f*)(in + i);
        v2u w;
        w.x = pack2_bf16(x[0], x[1]);
        w.y = pack2_bf16(x[2], x[3]);
        *(v2u*)(out + i) = w;
    }
}

// ---------------- WMMA GEMM:  C[M,N] = A[M,K] * W[N,K]^T ----------------
// MODE 0: bf16 row-major out.  MODE 1: f32 row-major out.
// MODE 2: bf16 transposed "Vt" out: Vt[((b*H + h)*64 + n)*T_DIM + s].
template <int MODE>
__global__ void __launch_bounds__(32, 1)
gemm_bf16(const unsigned short* __restrict__ A,
          const unsigned short* __restrict__ W,
          void* __restrict__ Cout,
          int M, int N, int K, float scale) {
    const int lane = threadIdx.x;
    const int grp  = lane >> 4;
    const int l16  = lane & 15;
    const int m0   = blockIdx.x * 32;
    const int n0   = blockIdx.y * 64;

    v8f acc0[4] = {}, acc1[4] = {};
    const unsigned short* arow0 = A + (long)(m0 + l16) * K;
    const unsigned short* arow1 = A + (long)(m0 + 16 + l16) * K;

    for (int kk = 0; kk < K; kk += 32) {
        __builtin_prefetch(arow0 + kk + 256, 0, 0);
        __builtin_prefetch(arow1 + kk + 256, 0, 0);
        v16bf af0 = *(const v16bf*)(arow0 + kk + grp * 16);
        v16bf af1 = *(const v16bf*)(arow1 + kk + grp * 16);
#pragma unroll
        for (int c = 0; c < 4; ++c) {
            const unsigned short* wrow =
                W + (long)(n0 + c * 16 + l16) * K + kk + grp * 16;
            v16bf bfrag = *(const v16bf*)wrow;
            acc0[c] = wmma_bf16(af0, bfrag, acc0[c]);
            acc1[c] = wmma_bf16(af1, bfrag, acc1[c]);
        }
    }

    if (MODE == 2) {
        // transposed store: col -> (h, n), row -> (b, s); 8 rows contiguous in s
        const long r0 = m0 + grp * 8;
        const long b  = r0 >> 11;          // / T_DIM
        const long s  = r0 & (T_DIM - 1);
#pragma unroll
        for (int c = 0; c < 4; ++c) {
            const long col = n0 + c * 16 + l16;
            const long h   = col >> 6;
            const long n   = col & 63;
            unsigned short* dst = (unsigned short*)Cout
                + ((b * NHEADS + h) * 64 + n) * T_DIM + s;
            struct { unsigned int w[4]; } t0, t1;
#pragma unroll
            for (int j = 0; j < 4; ++j) {
                t0.w[j] = pack2_bf16(acc0[c][2*j] * scale, acc0[c][2*j+1] * scale);
                t1.w[j] = pack2_bf16(acc1[c][2*j] * scale, acc1[c][2*j+1] * scale);
            }
            *(v8bf*)(dst)      = __builtin_bit_cast(v8bf, t0);
            *(v8bf*)(dst + 16) = __builtin_bit_cast(v8bf, t1);
        }
        return;
    }

#pragma unroll
    for (int c = 0; c < 4; ++c) {
#pragma unroll
        for (int j = 0; j < 8; ++j) {
            long row = m0 + grp * 8 + j;
            long col = n0 + c * 16 + l16;
            float v0 = acc0[c][j] * scale;
            float v1 = acc1[c][j] * scale;
            if (MODE == 1) {
                ((float*)Cout)[row * N + col]        = v0;
                ((float*)Cout)[(row + 16) * N + col] = v1;
            } else {
                ((unsigned short*)Cout)[row * N + col]        = f32_to_bf16(v0);
                ((unsigned short*)Cout)[(row + 16) * N + col] = f32_to_bf16(v1);
            }
        }
    }
}

// ---------------- fused differential flash attention ----------------
// grid = (T/16, B*H); 1 wave per block owns 16 query rows of one (b,h).
// Q is pre-scaled by (hd^-0.5 * log2(e)) so scores are in log2 domain and
// the online softmax uses raw v_exp (exp2). All hot global loads use
// SGPR-base + 32-bit VGPR offset addressing.
__global__ void __launch_bounds__(32, 1)
diffattn_kernel(const unsigned short* __restrict__ Q,
                const unsigned short* __restrict__ Km,
                const unsigned short* __restrict__ Vt,
                const float* __restrict__ lq1,
                const float* __restrict__ lk1,
                const float* __restrict__ lq2,
                const float* __restrict__ lk2,
                const float* __restrict__ subln,
                unsigned short* __restrict__ Out) {
    __shared__ float          ldsS[16][32];        // fp32 score tile (one head)
    __shared__ float          ldsR[2][16];         // per-row broadcast
    __shared__ unsigned short ldsP[2][16][32];     // bf16 P tiles, both heads

    const int lane = threadIdx.x;
    const int grp  = lane >> 4;
    const int l16  = lane & 15;
    const int t0   = blockIdx.x * 16;
    const int bh   = blockIdx.y;
    const int b    = bh >> 4;
    const int h    = bh & 15;

    // ---- lambda
    float p1 = lq1[lane] * lk1[lane];
    float p2 = lq2[lane] * lk2[lane];
#pragma unroll
    for (int m = 16; m >= 1; m >>= 1) {
        p1 += __shfl_xor(p1, m, 32);
        p2 += __shfl_xor(p2, m, 32);
    }
    const float lam = __expf(p1) - __expf(p2) + LAMBDA_INIT_F;

    // ---- Q fragments (A layout) for both differential heads
    const long qbase = (long)(b * T_DIM + t0 + l16) * EMB + (2 * h) * HD;
    v16bf q1 = *(const v16bf*)(Q + qbase + grp * 16);
    v16bf q2 = *(const v16bf*)(Q + qbase + HD + grp * 16);

    v8f O1[4] = {}, O2[4] = {};
    float mR[2] = {-1e30f, -1e30f};
    float lR[2] = {0.f, 0.f};

    // uniform bases; all per-lane offsets fit in 32 bits
    const unsigned short* kbase  = Km + (long)b * T_DIM * EMB;
    const unsigned short* vslice = Vt + (long)(b * NHEADS + h) * 64 * T_DIM;
    const int hoff0 = (2 * h) * HD + grp * 16;
    int vcol[4];
#pragma unroll
    for (int c = 0; c < 4; ++c) vcol[c] = (c * 16 + l16) * T_DIM;

    const v8f zero = {};
    int srow = l16 * EMB;          // (s0 + l16) * EMB, incremented per step
    int voff = grp * 16;           // s0 + grp*16, incremented per step

    for (int s0 = 0; s0 < T_DIM; s0 += 32, srow += 32 * EMB, voff += 32) {
#pragma unroll
        for (int head = 0; head < 2; ++head) {
            const int ka = srow + hoff0 + head * HD;
            v16bf kA = *(const v16bf*)(kbase + ka);
            v16bf kB = *(const v16bf*)(kbase + ka + 16 * EMB);
            v16bf qf = head ? q2 : q1;
            v8f Sa = wmma_bf16(qf, kA, zero);
            v8f Sb = wmma_bf16(qf, kB, zero);
            // scatter C layout -> row-major LDS
#pragma unroll
            for (int j = 0; j < 8; ++j) {
                ldsS[grp * 8 + j][l16]      = Sa[j];
                ldsS[grp * 8 + j][16 + l16] = Sb[j];
            }
            __syncthreads();
            // lane owns row l16, columns grp*16 .. +15 (scores in log2 domain)
            float sv[16];
#pragma unroll
            for (int i = 0; i < 4; ++i)
                *(v4f*)&sv[i * 4] = *(const v4f*)&ldsS[l16][grp * 16 + i * 4];
            float lm = sv[0];
#pragma unroll
            for (int i = 1; i < 16; ++i) lm = fmaxf(lm, sv[i]);
            lm = fmaxf(lm, __shfl_xor(lm, 16, 32));
            float mn = fmaxf(mR[head], lm);
            float al = exp2f(mR[head] - mn);
            mR[head] = mn;
            float ev[16];
            float ls = 0.f;
#pragma unroll
            for (int i = 0; i < 16; ++i) { ev[i] = exp2f(sv[i] - mn); ls += ev[i]; }
            ls += __shfl_xor(ls, 16, 32);
            lR[head] = lR[head] * al + ls;
            if (grp == 0) ldsR[head][l16] = al;
            struct { unsigned int w[8]; } pw;
#pragma unroll
            for (int i = 0; i < 8; ++i)
                pw.w[i] = pack2_bf16(ev[2 * i], ev[2 * i + 1]);
            *(v16bf*)&ldsP[head][l16][grp * 16] = __builtin_bit_cast(v16bf, pw);
            __syncthreads();
            // rescale accumulators: element j of av == alpha of row grp*8+j
            v8f av = *(const v8f*)&ldsR[head][grp * 8];
            if (head == 0) {
#pragma unroll
                for (int c = 0; c < 4; ++c) O1[c] = O1[c] * av;
            } else {
#pragma unroll
                for (int c = 0; c < 4; ++c) O2[c] = O2[c] * av;
            }
        }

        // ---- PV accumulation: P A-fragments from LDS, V B-fragments are
        // contiguous 32B global loads (sgpr base + 32-bit voffset).
        v16bf pa1 = *(const v16bf*)&ldsP[0][l16][grp * 16];
        v16bf pa2 = *(const v16bf*)&ldsP[1][l16][grp * 16];
#pragma unroll
        for (int c = 0; c < 4; ++c) {
            v16bf vb = *(const v16bf*)(vslice + vcol[c] + voff);
            O1[c] = wmma_bf16(pa1, vb, O1[c]);
            O2[c] = wmma_bf16(pa2, vb, O2[c]);
        }
        // next iteration's LDS writes are ordered by the head-loop barriers
    }

    // ---- epilogue: combine heads, RMSNorm(64), affine, write bf16
    __syncthreads();
    if (grp == 0) {
        ldsR[0][l16] = 1.0f / lR[0];
        ldsR[1][l16] = lam / lR[1];
    }
    __syncthreads();
    v8f inv1 = *(const v8f*)&ldsR[0][grp * 8];
    v8f inv2 = *(const v8f*)&ldsR[1][grp * 8];
    const float oscale = 1.0f - LAMBDA_INIT_F;
    float wsc[4];
#pragma unroll
    for (int c = 0; c < 4; ++c) wsc[c] = subln[c * 16 + l16] * oscale;
#pragma unroll
    for (int j = 0; j < 8; ++j) {
        float o[4];
        float ss = 0.f;
#pragma unroll
        for (int c = 0; c < 4; ++c) {
            o[c] = O1[c][j] * inv1[j] - O2[c][j] * inv2[j];
            ss += o[c] * o[c];
        }
#pragma unroll
        for (int m = 8; m >= 1; m >>= 1) ss += __shfl_xor(ss, m, 32);
        float r = rsqrtf(ss * (1.0f / 64.0f) + 1e-5f);
        long row = t0 + grp * 8 + j;
        long obase = (long)(b * T_DIM + row) * EMB + h * 64;
#pragma unroll
        for (int c = 0; c < 4; ++c) {
            float val = o[c] * r * wsc[c];
            Out[obase + c * 16 + l16] = f32_to_bf16(val);
        }
    }
}

// ---------------- launch ----------------
extern "C" void kernel_launch(void* const* d_in, const int* in_sizes, int n_in,
                              void* d_out, int out_size, void* d_ws, size_t ws_size,
                              hipStream_t stream) {
    const float* query = (const float*)d_in[0];
    const float* Wq    = (const float*)d_in[1];
    const float* Wk    = (const float*)d_in[2];
    const float* Wv    = (const float*)d_in[3];
    const float* Wo    = (const float*)d_in[4];
    const float* lq1   = (const float*)d_in[5];
    const float* lk1   = (const float*)d_in[6];
    const float* lq2   = (const float*)d_in[7];
    const float* lk2   = (const float*)d_in[8];
    const float* subln = (const float*)d_in[9];
    float* out = (float*)d_out;

    const size_t ACT = (size_t)M_ROWS * EMB * 2;   // 8 MiB
    const size_t WGT = (size_t)EMB * EMB * 2;      // 2 MiB
    char* ws = (char*)d_ws;
    unsigned short* Xb  = (unsigned short*)(ws);
    unsigned short* Wqb = (unsigned short*)(ws + ACT);
    unsigned short* Wkb = (unsigned short*)(ws + ACT + WGT);
    unsigned short* Wvb = (unsigned short*)(ws + ACT + 2 * WGT);
    unsigned short* Wob = (unsigned short*)(ws + ACT + 3 * WGT);
    unsigned short* Qb  = (unsigned short*)(ws + ACT + 4 * WGT);
    unsigned short* Kb  = (unsigned short*)(ws + 2 * ACT + 4 * WGT);
    unsigned short* Vtb = (unsigned short*)(ws + 3 * ACT + 4 * WGT);
    unsigned short* Ab  = (unsigned short*)(ws + 4 * ACT + 4 * WGT);

    {
        int n = M_ROWS * EMB;
        cast_f32_bf16<<<(n / 4 + 255) / 256, 256, 0, stream>>>(query, Xb, n);
        int nw = EMB * EMB;
        cast_f32_bf16<<<(nw / 4 + 255) / 256, 256, 0, stream>>>(Wq, Wqb, nw);
        cast_f32_bf16<<<(nw / 4 + 255) / 256, 256, 0, stream>>>(Wk, Wkb, nw);
        cast_f32_bf16<<<(nw / 4 + 255) / 256, 256, 0, stream>>>(Wv, Wvb, nw);
        cast_f32_bf16<<<(nw / 4 + 255) / 256, 256, 0, stream>>>(Wo, Wob, nw);
    }

    dim3 ggrid(M_ROWS / 32, EMB / 64);
    // Q pre-scaled into log2 domain for the exp2-based online softmax
    gemm_bf16<0><<<ggrid, 32, 0, stream>>>(Xb, Wqb, Qb, M_ROWS, EMB, EMB,
                                           Q_SCALE * LOG2E_F);
    gemm_bf16<0><<<ggrid, 32, 0, stream>>>(Xb, Wkb, Kb, M_ROWS, EMB, EMB, 1.0f);
    gemm_bf16<2><<<ggrid, 32, 0, stream>>>(Xb, Wvb, Vtb, M_ROWS, EMB, EMB, 1.0f);

    dim3 agrid(T_DIM / 16, BATCH * NHEADS);
    diffattn_kernel<<<agrid, 32, 0, stream>>>(Qb, Kb, Vtb, lq1, lk1, lq2, lk2, subln, Ab);

    gemm_bf16<1><<<ggrid, 32, 0, stream>>>(Ab, Wob, out, M_ROWS, EMB, EMB, 1.0f);
}
